// MultiScaleRetention_54709293417114
// MI455X (gfx1250) — matrix operand order repaired and checked by reading, not compiled, not run on hardware
//
#include <hip/hip_runtime.h>
#include <cstdint>
#include <cstddef>

// ---------------------------------------------------------------------------
// MultiScaleRetention for MI455X (gfx1250), wave32 + WMMA bf16.
// B=2, S=2048, H=16, D_HEAD=64, HID=1024.
// ---------------------------------------------------------------------------

#define BB   2
#define SS   2048
#define HH   16
#define DH   64
#define HID  1024
#define MTOT (BB * SS)      // 4096 rows for the dense GEMMs

typedef __attribute__((ext_vector_type(16))) __bf16 v16bf;
typedef __attribute__((ext_vector_type(8)))  float  v8f;

// ---- workspace layout (bytes) ----------------------------------------------
#define XBF_OFF  (size_t)0                      // x as bf16              8 MB
#define W1T_OFF  (XBF_OFF + (size_t)MTOT*HID*2) // W1^T bf16              2 MB
#define W2T_OFF  (W1T_OFF + (size_t)HID*HID*2)  // W2^T bf16              2 MB
#define QC_OFF   (W2T_OFF + (size_t)HID*HID*2)  // [b,h,s,128] q*cs||q*sn 16 MB
#define KC_OFF   (QC_OFF  + (size_t)BB*HH*SS*128*2)
#define VT_OFF   (KC_OFF  + (size_t)BB*HH*SS*128*2) // V^T [b,h,d,s]      8 MB
#define RET_OFF  (VT_OFF  + (size_t)BB*HH*DH*SS*2)  // ret f32 [b,s,hid] 16 MB
#define UBF_OFF  (RET_OFF + (size_t)MTOT*HID*4)     // gated act bf16     8 MB

// ---------------------------------------------------------------------------
// WMMA operand loaders, both using the same contiguous k order (k = half*16 + j).
// A-matrix 16x32 (MxK), src row-major with stride lda:
//   lane m = l&15, half = l>>4; elem j -> k = half*16 + j  (j=0..15)
// Both runs are 8 contiguous bf16 -> vectorizes to 16B loads.
__device__ __forceinline__ v16bf load_a16x32(const __bf16* base, int lda, int lane) {
    const int m  = lane & 15;
    const int hf = lane >> 4;
    const __bf16* p0 = base + (size_t)m * lda + hf * 16;
    const __bf16* p1 = p0 + 8;
    v16bf a;
#pragma unroll
    for (int j = 0; j < 8; ++j) { a[j] = p0[j]; a[8 + j] = p1[j]; }
    return a;
}

// B-matrix 32x16 (KxN) from K-contiguous ("transposed") storage:
//   element (k,n) lives at base[n*ldb + k], with k = half*16 + j  (j=0..15).
// 16 contiguous bf16 per lane -> vectorizes to 32B of loads.
__device__ __forceinline__ v16bf load_bT32x16(const __bf16* base, int ldb, int lane) {
    const int n  = lane & 15;
    const int hf = lane >> 4;
    const __bf16* p = base + (size_t)n * ldb + hf * 16;
    v16bf b;
#pragma unroll
    for (int j = 0; j < 16; ++j) b[j] = p[j];
    return b;
}

__device__ __forceinline__ v8f wmma_bf16(v16bf a, v16bf b, v8f c) {
    // D = A(16x32) * B(32x16) + C, f32 accumulate
    return __builtin_amdgcn_wmma_f32_16x16x32_bf16(
        /*neg_a=*/false, a, /*neg_b=*/false, b,
        /*c_mod=*/(short)0, c, /*reuse_a=*/false, /*reuse_b=*/false);
}

// ---------------------------------------------------------------------------
// Prep kernels
__global__ void k_cvt_bf16(const float* __restrict__ src, __bf16* __restrict__ dst, int n) {
    int i = blockIdx.x * blockDim.x + threadIdx.x;
    if (i < n) dst[i] = (__bf16)src[i];
}

// wt[n*HID + k] = w[k*HID + n]  (K-contiguous B operand for out = act @ W)
__global__ void k_transpose_bf16(const float* __restrict__ w, __bf16* __restrict__ wt) {
    int i = blockIdx.x * blockDim.x + threadIdx.x;   // over HID*HID
    int n = i >> 10, k = i & (HID - 1);
    wt[(size_t)n * HID + k] = (__bf16)w[(size_t)k * HID + n];
}

// ---------------------------------------------------------------------------
// Fused per-head QKV projection + rotary. One thread per (b,h,s).
// qc/kc: [b,h,s,128] = concat(q*cos, q*sin) / concat(k*cos, k*sin), bf16
// vt:    [b,h,d,s]  (transposed so retention B-tiles are K-contiguous), bf16
__global__ void k_qkv_rotary(const float* __restrict__ x,
                             const float* __restrict__ wq, const float* __restrict__ bq,
                             const float* __restrict__ wk, const float* __restrict__ bk,
                             const float* __restrict__ wv, const float* __restrict__ theta,
                             __bf16* __restrict__ qc, __bf16* __restrict__ kc,
                             __bf16* __restrict__ vt) {
    int idx = blockIdx.x * blockDim.x + threadIdx.x;   // B*H*S threads
    int s = idx & (SS - 1);
    int h = (idx >> 11) & (HH - 1);
    int b = idx >> 15;

    const float* xr = x + ((size_t)(b * SS + s)) * HID + h * DH;
    const float* Wq = wq + (size_t)h * DH * DH;
    const float* Wk = wk + (size_t)h * DH * DH;
    const float* Wv = wv + (size_t)h * DH * DH;

    float xv[DH];
#pragma unroll 8
    for (int d = 0; d < DH; ++d) xv[d] = xr[d];

    size_t qko = ((size_t)((b * HH + h) * SS + s)) * 128;
    size_t vto = ((size_t)(b * HH + h)) * DH * SS + s;

    for (int e = 0; e < DH; ++e) {
        float aq = bq[h * DH + e];
        float ak = bk[h * DH + e];
        float av = 0.f;
#pragma unroll 8
        for (int d = 0; d < DH; ++d) {
            float xd = xv[d];
            aq += xd * Wq[d * DH + e];
            ak += xd * Wk[d * DH + e];
            av += xd * Wv[d * DH + e];
        }
        float ang = (float)s * theta[h * DH + e];
        float sn, cs;
        __sincosf(ang, &sn, &cs);
        qc[qko + e]      = (__bf16)(aq * cs);
        qc[qko + 64 + e] = (__bf16)(aq * sn);
        kc[qko + e]      = (__bf16)(ak * cs);
        kc[qko + 64 + e] = (__bf16)(ak * sn);
        vt[vto + (size_t)e * SS] = (__bf16)av;
    }
}

// ---------------------------------------------------------------------------
// Flash-style retention. grid = (B*H, S/128); block = 256 = 8 waves.
// Each wave owns a 16-row query strip; walks key chunks FROM THE DIAGONAL
// BACKWARD so the decay factors gamma^(s-t) update with one multiply by
// gamma^32 per chunk (expf + causal mask only on the 2 diagonal chunks).
// Scores never touch HBM: QK^T in WMMA accumulators -> decay in registers ->
// bf16 P tile via per-wave LDS relayout -> P@V WMMA.
__global__ void k_retention(const __bf16* __restrict__ qc, const __bf16* __restrict__ kc,
                            const __bf16* __restrict__ vt, float* __restrict__ ret) {
    __shared__ __bf16 plds[8][16 * 32];          // 8 KB: per-wave 16x32 P tile

    const int lane = threadIdx.x & 31;
    const int wave = threadIdx.x >> 5;
    const int bh   = blockIdx.x;                 // b*H + h
    const int h    = bh & (HH - 1);
    const int qb   = blockIdx.y * 128 + wave * 16;

    // decay: gamma_h = 1 - exp(linspace(log(1/512), log(1/32), H))
    const float lo = -6.23832462503950778f;      // log(1/512)
    const float hi = -3.46573590279972655f;      // log(1/32)
    const float gamma = 1.f - __expf(lo + (hi - lo) * ((float)h / (float)(HH - 1)));
    const float logg  = __logf(gamma);
    const float g32   = __expf(32.f * logg);     // gamma^32 per-chunk update

    const __bf16* qbase = qc + (size_t)bh * SS * 128;
    const __bf16* kbase = kc + (size_t)bh * SS * 128;
    const __bf16* vbase = vt + (size_t)bh * DH * SS;

    const int nCol = lane & 15;
    const int hf   = lane >> 4;

    // Q strip A-tiles (16x32 each over the 128 rotary features) — reused all loop long
    v16bf aq[4];
#pragma unroll
    for (int f = 0; f < 4; ++f)
        aq[f] = load_a16x32(qbase + (size_t)qb * 128 + f * 32, 128, lane);

    v8f acc[4] = {};                             // O strip: 16 rows x 64 cols
    __bf16* pw = &plds[wave][0];

    float f0[8], f1[8];                          // running decay factors
    const int tcmax = ((qb + 15) >> 5) << 5;     // diagonal key chunk

    int it = 0;
    for (int tc = tcmax; tc >= 0; tc -= 32, ++it) {
        // ---- S = Q K^T over 128 features (K stored row-major => K-contiguous B)
        v8f s0 = {}, s1 = {};
        const __bf16* kb0 = kbase + (size_t)tc * 128;
        const __bf16* kb1 = kbase + (size_t)(tc + 16) * 128;
#pragma unroll
        for (int f = 0; f < 4; ++f) {
            v16bf b0 = load_bT32x16(kb0 + f * 32, 128, lane);
            s0 = wmma_bf16(aq[f], b0, s0);
            v16bf b1 = load_bT32x16(kb1 + f * 32, 128, lane);
            s1 = wmma_bf16(aq[f], b1, s1);
        }

        // ---- decay factors: expf+mask on diagonal chunks, then *= gamma^32
        if (it < 2) {
#pragma unroll
            for (int r = 0; r < 8; ++r) {
                int sa = qb + hf * 8 + r;            // absolute query position
                int d0 = sa - (tc + nCol);
                f0[r] = (d0 >= 0) ? __expf((float)d0 * logg) : 0.f;
                int d1 = d0 - 16;
                f1[r] = (d1 >= 0) ? __expf((float)d1 * logg) : 0.f;
            }
        } else {
#pragma unroll
            for (int r = 0; r < 8; ++r) { f0[r] *= g32; f1[r] *= g32; }
        }

        // ---- apply decay, relayout C->A via per-wave LDS tile
#pragma unroll
        for (int r = 0; r < 8; ++r) {
            pw[(hf * 8 + r) * 32 + nCol]      = (__bf16)(s0[r] * f0[r]);
            pw[(hf * 8 + r) * 32 + 16 + nCol] = (__bf16)(s1[r] * f1[r]);
        }
        asm volatile("s_wait_dscnt 0" ::: "memory");   // wave-private LDS RAW
        v16bf pa = load_a16x32(pw, 32, lane);

        // ---- O += P (16x32) @ V (32x64), V^T storage => K-contiguous B tiles
#pragma unroll
        for (int dslice = 0; dslice < 4; ++dslice) {
            v16bf bv = load_bT32x16(vbase + (size_t)(dslice * 16) * SS + tc, SS, lane);
            acc[dslice] = wmma_bf16(pa, bv, acc[dslice]);
        }
    }

    // ---- write ret f32 [b, s, h*64 + d]
    const int b = bh >> 4;
    float* ro = ret + (size_t)b * SS * HID + h * DH;
#pragma unroll
    for (int r = 0; r < 8; ++r) {
        int sa = qb + hf * 8 + r;
        float* row = ro + (size_t)sa * HID + nCol;
        row[0]  = acc[0][r];
        row[16] = acc[1][r];
        row[32] = acc[2][r];
        row[48] = acc[3][r];
    }
}

// ---------------------------------------------------------------------------
// In-place GroupNorm (groups = heads, 64 channels/group). One thread per group.
__global__ void k_groupnorm(float* __restrict__ ret, const float* __restrict__ gw,
                            const float* __restrict__ gb) {
    int g = blockIdx.x * blockDim.x + threadIdx.x;   // B*S*H groups
    int h = g & (HH - 1);
    float* p = ret + (size_t)(g >> 4) * HID + h * DH;

    float v[DH];
    float mean = 0.f;
#pragma unroll 8
    for (int i = 0; i < DH; ++i) { v[i] = p[i]; mean += v[i]; }
    mean *= (1.f / DH);
    float var = 0.f;
#pragma unroll 8
    for (int i = 0; i < DH; ++i) { float d = v[i] - mean; var += d * d; }
    var *= (1.f / DH);
    float rs = __frsqrt_rn(var + 1e-5f);
    const float* w = gw + h * DH;
    const float* bb = gb + h * DH;
#pragma unroll 8
    for (int i = 0; i < DH; ++i) p[i] = (v[i] - mean) * rs * w[i] + bb[i];
}

// ---------------------------------------------------------------------------
// Dense GEMM core with cooperative double-buffered LDS staging of the shared
// 64x32 W^T tile (all 8 waves consume the same B tile -> stage once per block
// instead of 8 redundant L2 streams). 256 threads each move one 16B chunk.
// grid = (M/128, N/64); each wave computes a 16x64 strip, K=1024.
__device__ __forceinline__ void stage_btile(__bf16* dst, const __bf16* __restrict__ bt,
                                            int nb, int kc, int tid) {
    const int n = tid >> 2;          // 0..63
    const int c = tid & 3;           // 16B chunk within the 32-elem row
    const uint4 v = *(const uint4*)(bt + (size_t)(nb + n) * HID + kc + c * 8);
    *(uint4*)(dst + n * 32 + c * 8) = v;
}

template <int FUSE_GATE>
__device__ __forceinline__ void gemm_core(const __bf16* __restrict__ a,
                                          const __bf16* __restrict__ bt,
                                          const float* __restrict__ retn,
                                          __bf16* __restrict__ u,
                                          float* __restrict__ out) {
    __shared__ __bf16 bstage[2][64 * 32];        // 2 x 4KB double buffer

    const int tid  = threadIdx.x;
    const int lane = tid & 31;
    const int wave = tid >> 5;
    const int mb = blockIdx.x * 128 + wave * 16;
    const int nb = blockIdx.y * 64;

    stage_btile(&bstage[0][0], bt, nb, 0, tid);  // prologue

    v8f acc[4] = {};
    for (int kc = 0; kc < HID; kc += 32) {
        const int cur = (kc >> 5) & 1;
        __syncthreads();                         // publishes buf[cur]; retires reads of buf[1-cur]
        if (kc + 32 < HID) {
            stage_btile(&bstage[1 - cur][0], bt, nb, kc + 32, tid);  // overlap with compute
            __builtin_prefetch(a + (size_t)mb * HID + kc + 32, 0, 0);
        }
        v16bf av = load_a16x32(a + (size_t)mb * HID + kc, HID, lane);
#pragma unroll
        for (int sN = 0; sN < 4; ++sN) {
            v16bf bv = load_bT32x16(&bstage[cur][sN * 16 * 32], 32, lane);
            acc[sN] = wmma_bf16(av, bv, acc[sN]);
        }
    }

    const int nCol = lane & 15, hf = lane >> 4;
#pragma unroll
    for (int sN = 0; sN < 4; ++sN) {
#pragma unroll
        for (int r = 0; r < 8; ++r) {
            int m = mb + hf * 8 + r;
            int n = nb + sN * 16 + nCol;
            float g = acc[sN][r];
            if (FUSE_GATE) {
                float sw = g / (1.f + __expf(-g));   // g * sigmoid(g)
                u[(size_t)m * HID + n] = (__bf16)(sw + retn[(size_t)m * HID + n]);
            } else {
                out[(size_t)m * HID + n] = g;
            }
        }
    }
}

// GEMM1: u = swish(x @ W1) + retn, bf16 out.
__global__ void k_gemm_gate(const __bf16* __restrict__ a, const __bf16* __restrict__ bt,
                            const float* __restrict__ retn, __bf16* __restrict__ u) {
    gemm_core<1>(a, bt, retn, u, nullptr);
}

// GEMM2: out = u @ W2, f32 out.
__global__ void k_gemm_out(const __bf16* __restrict__ a, const __bf16* __restrict__ bt,
                           float* __restrict__ out) {
    gemm_core<0>(a, bt, nullptr, nullptr, out);
}

// ---------------------------------------------------------------------------
extern "C" void kernel_launch(void* const* d_in, const int* in_sizes, int n_in,
                              void* d_out, int out_size, void* d_ws, size_t ws_size,
                              hipStream_t stream) {
    (void)in_sizes; (void)n_in; (void)out_size; (void)ws_size;

    const float* x     = (const float*)d_in[0];
    const float* wq    = (const float*)d_in[1];
    const float* bq    = (const float*)d_in[2];
    const float* wk    = (const float*)d_in[3];
    const float* bk    = (const float*)d_in[4];
    const float* wv    = (const float*)d_in[5];
    const float* theta = (const float*)d_in[6];
    const float* gnw   = (const float*)d_in[7];
    const float* gnb   = (const float*)d_in[8];
    const float* w1    = (const float*)d_in[9];
    const float* w2    = (const float*)d_in[10];

    char* ws = (char*)d_ws;
    __bf16* xbf = (__bf16*)(ws + XBF_OFF);
    __bf16* w1t = (__bf16*)(ws + W1T_OFF);
    __bf16* w2t = (__bf16*)(ws + W2T_OFF);
    __bf16* qcb = (__bf16*)(ws + QC_OFF);
    __bf16* kcb = (__bf16*)(ws + KC_OFF);
    __bf16* vtb = (__bf16*)(ws + VT_OFF);
    float*  ret = (float*) (ws + RET_OFF);
    __bf16* ubf = (__bf16*)(ws + UBF_OFF);

    // 1. precision/layout prep
    k_cvt_bf16<<<(MTOT * HID + 255) / 256, 256, 0, stream>>>(x, xbf, MTOT * HID);
    k_transpose_bf16<<<(HID * HID) / 256, 256, 0, stream>>>(w1, w1t);
    k_transpose_bf16<<<(HID * HID) / 256, 256, 0, stream>>>(w2, w2t);

    // 2. fused QKV + rotary
    k_qkv_rotary<<<(BB * HH * SS) / 256, 256, 0, stream>>>(
        x, wq, bq, wk, bk, wv, theta, qcb, kcb, vtb);

    // 3. flash-style decayed retention (WMMA)
    dim3 rg(BB * HH, SS / 128);
    k_retention<<<rg, 256, 0, stream>>>(qcb, kcb, vtb, ret);

    // 4. per-head GroupNorm, in place
    k_groupnorm<<<(BB * SS * HH) / 256, 256, 0, stream>>>(ret, gnw, gnb);

    // 5. gate GEMM fused with swish + residual, then output GEMM (WMMA)
    dim3 gg(MTOT / 128, HID / 64);
    k_gemm_gate<<<gg, 256, 0, stream>>>(xbf, w1t, ret, ubf);
    k_gemm_out <<<gg, 256, 0, stream>>>(ubf, w2t, (float*)d_out);
}
